// MultiHeadSelfAttention_29506425323673
// MI455X (gfx1250) — compile-verified, matrix operand order checked by reading, not run on hardware
//
#include <hip/hip_runtime.h>

// -------- MI455X / gfx1250: wave32, WMMA bf16 16x16x32 (f32 accum) --------
// TDM (tensor_load_to_lds) double-buffered tile staging, ASYNC-free epilogues.

#define BATCH   4
#define SEQ     2048
#define DMODEL  1024
#define NHEADS  16
#define DHEAD   64

typedef __attribute__((ext_vector_type(16))) __bf16 v16bf;
typedef __attribute__((ext_vector_type(8)))  float  v8f;
typedef __attribute__((ext_vector_type(4)))  unsigned int v4u;
typedef __attribute__((ext_vector_type(8)))  int v8i;
typedef __attribute__((ext_vector_type(4)))  int v4i;
typedef unsigned short ushort_t;

#if defined(__AMDGCN__) && __has_builtin(__builtin_amdgcn_tensor_load_to_lds)
#define HAVE_TDM 1
#else
#define HAVE_TDM 0
#endif

// float -> bf16 (round to nearest even)
__device__ __forceinline__ ushort_t f2bf(float f) {
    unsigned int u = __float_as_uint(f);
    u += 0x7FFFu + ((u >> 16) & 1u);
    return (ushort_t)(u >> 16);
}

__device__ __forceinline__ v8f wmma_bf16(v16bf a, v16bf b, v8f c) {
    // D = A(16x32 bf16) * B(32x16 bf16) + C(16x16 f32)
    return __builtin_amdgcn_wmma_f32_16x16x32_bf16(
        false, a, false, b, (short)0, c, false, false);
}

// A/B fragment (16x32 bf16) from row-major bf16 memory (global or LDS).
__device__ __forceinline__ v16bf frag_b16(const ushort_t* p, int stride) {
    int lane = threadIdx.x & 31;
    const ushort_t* q = p + (lane & 15) * stride + ((lane >> 4) << 3);
    union { uint4 u[2]; v16bf v; } t;
    t.u[0] = *(const uint4*)(q);
    t.u[1] = *(const uint4*)(q + 16);
    return t.v;
}

// 32-bit LDS byte address of a shared-memory object (low bits of generic addr)
__device__ __forceinline__ unsigned int lds_u32(const void* p) {
    return (unsigned int)(unsigned long long)p;
}

// Issue a TDM 2D tile DMA: global (row-major) -> LDS, tracked by TENSORcnt.
// elem_code: 0=1B,1=2B,2=4B. dims/strides in elements. Tile = tile1 rows x tile0.
__device__ __forceinline__ void tdm_load_2d(
    const void* gsrc, unsigned int lds_addr, unsigned int elem_code,
    unsigned int dim0, unsigned int dim1, unsigned long long stride0,
    unsigned int tile0, unsigned int tile1)
{
#if HAVE_TDM
    unsigned long long ga = (unsigned long long)gsrc;
    v4u g0;
    g0[0] = 1u;                                           // count=1 (valid user D#)
    g0[1] = lds_addr;                                     // lds_addr [63:32]
    g0[2] = (unsigned int)ga;                             // global_addr[31:0]
    g0[3] = (unsigned int)((ga >> 32) & 0x01FFFFFFu)      // global_addr[56:32]
          | 0x80000000u;                                  // type=2 ("image") [127:126]
    v8i g1;
    g1[0] = (int)(elem_code << 16);                       // data_size [17:16]
    g1[1] = (int)(dim0 << 16);                            // tensor_dim0[15:0]  -> [63:48]
    g1[2] = (int)((dim0 >> 16) | (dim1 << 16));           // dim0[31:16], dim1[15:0]
    g1[3] = (int)((dim1 >> 16) | (tile0 << 16));          // dim1[31:16], tile_dim0
    g1[4] = (int)tile1;                                   // tile_dim1 (tile_dim2=0)
    g1[5] = (int)(stride0 & 0xFFFFFFFFull);               // tensor_dim0_stride[31:0]
    g1[6] = (int)((stride0 >> 32) & 0xFFFFull);           // stride0[47:32]
    g1[7] = 0;
    v4i z4 = {0, 0, 0, 0};
    v8i z8 = {0, 0, 0, 0, 0, 0, 0, 0};
    __builtin_amdgcn_tensor_load_to_lds(g0, g1, z4, z4, z8, 0);
#else
    (void)gsrc; (void)lds_addr; (void)elem_code; (void)dim0; (void)dim1;
    (void)stride0; (void)tile0; (void)tile1;
#endif
}

__device__ __forceinline__ void tdm_wait() {
#if HAVE_TDM && __has_builtin(__builtin_amdgcn_s_wait_tensorcnt)
    __builtin_amdgcn_s_wait_tensorcnt(0);
#endif
}

// ---------------------------------------------------------------------------
// One-shot f32 -> bf16 conversion (x and weights) so hot loops are pure WMMA.
// ---------------------------------------------------------------------------
__global__ __launch_bounds__(256) void cvt_kernel(const float* __restrict__ src,
                                                  ushort_t* __restrict__ dst, int n) {
    for (int i = blockIdx.x * 256 + threadIdx.x; i < n; i += gridDim.x * 256)
        dst[i] = f2bf(src[i]);
}

// ---------------------------------------------------------------------------
// GEMM: out = A(8192x1024 bf16) @ W^T(bf16) + bias, TDM double-buffered W tiles
//   MODE 0: fused RoPE, bf16 out [B,H,T,DHEAD]  (Q/K; scale folded into Q)
//   MODE 1: no RoPE,    bf16 out [B,H,T,DHEAD]  (V)
//   MODE 2: f32 out [M,1024]                    (output projection)
// Block = 256 thr (8 waves). Tile: 128 rows x 64 cols; wave = 16 rows x 64.
// ---------------------------------------------------------------------------
template <int MODE>
__global__ __launch_bounds__(256) void gemm_kernel(
    const ushort_t* __restrict__ Ab,        // bf16 A [8192,1024]
    const ushort_t* __restrict__ Wb,        // bf16 W [1024,1024]
    const float* __restrict__ bias,         // f32 [1024]
    ushort_t* __restrict__ out_bf,          // bf16 [B,H,T,DHEAD]
    float* __restrict__ out_f32,            // f32 [M,1024]
    float out_scale)
{
    const int tid  = threadIdx.x;
    const int wave = tid >> 5;
    const int lane = tid & 31;
    const int nbase = blockIdx.y * 64;
    const int mwave = blockIdx.x * 128 + wave * 16;

    __shared__ __align__(16) ushort_t wtile[2][64 * 32]; // double-buffered W tiles
    __shared__ float bias_s[64];
    __shared__ float ctile[8][16 * 64];                  // per-wave epilogue stage

    if (tid < 64) bias_s[tid] = bias[nbase + tid];

    // ---- prologue: stage W tile 0 ----
#if HAVE_TDM
    if (wave == 0) {
        tdm_load_2d(Wb + (size_t)nbase * DMODEL, lds_u32(&wtile[0][0]),
                    1, DMODEL, DMODEL, DMODEL, 32, 64);
        tdm_wait();
    }
#else
    {
        int n = tid >> 2, k0 = (tid & 3) << 3;
        const uint4* wp = (const uint4*)(Wb + (size_t)(nbase + n) * DMODEL + k0);
        *(uint4*)&wtile[0][n * 32 + k0] = wp[0];
    }
#endif
    __syncthreads();

    v8f acc[4] = {};
    for (int kc = 0; kc < DMODEL; kc += 32) {
        const int cur = (kc >> 5) & 1, nxt = cur ^ 1;
        if (kc + 32 < DMODEL) {
#if HAVE_TDM
            if (wave == 0)
                tdm_load_2d(Wb + (size_t)nbase * DMODEL + kc + 32,
                            lds_u32(&wtile[nxt][0]), 1, DMODEL, DMODEL, DMODEL, 32, 64);
#else
            int n = tid >> 2, k0 = (tid & 3) << 3;
            const uint4* wp = (const uint4*)(Wb + (size_t)(nbase + n) * DMODEL + kc + 32 + k0);
            *(uint4*)&wtile[nxt][n * 32 + k0] = wp[0];
#endif
        }
        v16bf afrag = frag_b16(Ab + (size_t)mwave * DMODEL + kc, DMODEL);
#pragma unroll
        for (int nt = 0; nt < 4; ++nt) {
            v16bf bfrag = frag_b16(&wtile[cur][nt * 16 * 32], 32);
            acc[nt] = wmma_bf16(afrag, bfrag, acc[nt]);
        }
#if HAVE_TDM
        if (wave == 0) tdm_wait();
#endif
        __syncthreads();
    }

    if constexpr (MODE == 2) {
#pragma unroll
        for (int nt = 0; nt < 4; ++nt) {
            float b = bias_s[nt * 16 + (lane & 15)];
            int n = nbase + nt * 16 + (lane & 15);
#pragma unroll
            for (int j = 0; j < 8; ++j) {
                int m = mwave + j + ((lane >> 4) << 3);
                out_f32[(size_t)m * DMODEL + n] = acc[nt][j] + b;
            }
        }
    } else {
        // stage C (+bias) in per-wave LDS, then RoPE + [B,H,T,DH] bf16 store
        float* ct = ctile[wave];
#pragma unroll
        for (int nt = 0; nt < 4; ++nt) {
            float b = bias_s[nt * 16 + (lane & 15)];
#pragma unroll
            for (int j = 0; j < 8; ++j) {
                int row = j + ((lane >> 4) << 3);
                ct[row * 64 + nt * 16 + (lane & 15)] = acc[nt][j] + b;
            }
        }
        int r    = lane >> 1;           // row 0..15
        int half = lane & 1;            // column half (32 cols each)
        int token = mwave + r;
        int bidx  = token >> 11;        // / SEQ
        int t     = token & (SEQ - 1);
        int head  = blockIdx.y;         // N-tile == one head

        const float* src = ct + r * 64 + half * 32;
        alignas(16) ushort_t obuf[32];
#pragma unroll
        for (int i = 0; i < 16; ++i) {
            float v0 = src[2 * i], v1 = src[2 * i + 1];
            float o0, o1;
            if constexpr (MODE == 0) {
                int p = half * 16 + i;                               // rotary pair
                float inv = __expf(-0.28782313662425572f * (float)p); // 10000^(-2p/64)
                float ang = (float)t * inv;
                float c = __cosf(ang), s = __sinf(ang);
                o0 = v0 * c - v1 * s;
                o1 = v0 * s + v1 * c;
            } else {
                o0 = v0; o1 = v1;
            }
            obuf[2 * i]     = f2bf(o0 * out_scale);
            obuf[2 * i + 1] = f2bf(o1 * out_scale);
        }
        ushort_t* dst = out_bf +
            ((((size_t)bidx * NHEADS + head) * SEQ + t) * DHEAD + half * 32);
#pragma unroll
        for (int i = 0; i < 4; ++i) ((uint4*)dst)[i] = ((const uint4*)obuf)[i];
    }
}

// ---------------------------------------------------------------------------
// Flash attention: grid (T/128, B*H), 256 thr (8 waves, 16 queries each).
// TDM streams K tiles into double-buffered LDS while waves run WMMA; V tiles
// are loaded to registers pre-compute and transposed-scattered post-compute.
// ---------------------------------------------------------------------------
__global__ __launch_bounds__(256) void attn_kernel(
    const ushort_t* __restrict__ Q,   // bf16 [B,H,T,DH] (pre-scaled by 1/8)
    const ushort_t* __restrict__ K,   // bf16 [B,H,T,DH] (RoPE applied)
    const ushort_t* __restrict__ V,   // bf16 [B,H,T,DH]
    ushort_t* __restrict__ ctx)       // bf16 [B,T,DMODEL]
{
    const int tid  = threadIdx.x;
    const int wave = tid >> 5;
    const int lane = tid & 31;
    const int qblk = blockIdx.x;
    const int bh   = blockIdx.y;
    const int b    = bh >> 4;
    const int h    = bh & 15;
    const int qbase = qblk * 128 + wave * 16;
    const size_t base = (size_t)bh * SEQ * DHEAD;

    __shared__ __align__(16) ushort_t kt[2][64 * 64];   // [key][d], double buffer
    __shared__ __align__(16) ushort_t vt[2][64 * 64];   // [d][key], double buffer
    __shared__ __align__(16) ushort_t pt[8][16 * 64];   // per-wave P scratch

    const int r  = tid >> 2;
    const int c0 = (tid & 3) << 4;

    v16bf qf[2];
    qf[0] = frag_b16(Q + base + (size_t)qbase * DHEAD +  0, DHEAD);
    qf[1] = frag_b16(Q + base + (size_t)qbase * DHEAD + 32, DHEAD);

    v8f O[4] = {};
    float rmax[8], rsum[8];
#pragma unroll
    for (int j = 0; j < 8; ++j) { rmax[j] = -1e30f; rsum[j] = 0.0f; }

    // ---- prologue: stage K0 (TDM) and V0^T (register transpose) ----
#if HAVE_TDM
    if (wave == 0)
        tdm_load_2d(K + base, lds_u32(&kt[0][0]), 1, DHEAD, SEQ, DHEAD, 64, 64);
#else
    {
        const uint4* ks = (const uint4*)(K + base + (size_t)r * DHEAD + c0);
        ((uint4*)&kt[0][r * 64 + c0])[0] = ks[0];
        ((uint4*)&kt[0][r * 64 + c0])[1] = ks[1];
    }
#endif
    {
        union { uint4 u[2]; ushort_t s[16]; } vv;
        const uint4* vs = (const uint4*)(V + base + (size_t)r * DHEAD + c0);
        vv.u[0] = vs[0]; vv.u[1] = vs[1];
#pragma unroll
        for (int i = 0; i < 16; ++i) vt[0][(c0 + i) * 64 + r] = vv.s[i];
    }
#if HAVE_TDM
    if (wave == 0) tdm_wait();
#endif
    __syncthreads();

    const int kbmax = 2 * qblk + 1;
    for (int kb = 0; kb <= kbmax; ++kb) {
        const int cur = kb & 1, nxt = cur ^ 1;
        const bool has_next = kb < kbmax;

        union { uint4 u[2]; ushort_t s[16]; } vv = {};
        if (has_next) {
#if HAVE_TDM
            if (wave == 0)
                tdm_load_2d(K + base + (size_t)(kb + 1) * 64 * DHEAD,
                            lds_u32(&kt[nxt][0]), 1, DHEAD, SEQ, DHEAD, 64, 64);
#else
            const uint4* ks = (const uint4*)(K + base + (size_t)((kb + 1) * 64 + r) * DHEAD + c0);
            ((uint4*)&kt[nxt][r * 64 + c0])[0] = ks[0];
            ((uint4*)&kt[nxt][r * 64 + c0])[1] = ks[1];
#endif
            const uint4* vs = (const uint4*)(V + base + (size_t)((kb + 1) * 64 + r) * DHEAD + c0);
            vv.u[0] = vs[0]; vv.u[1] = vs[1];
            if (kb + 2 <= kbmax)   // CDNA5 global_prefetch two tiles ahead
                __builtin_prefetch(V + base + (size_t)(kb + 2) * 64 * DHEAD, 0, 1);
        }

        if (kb * 64 <= qbase + 15) {   // wave has unmasked keys in this block
            // ---- S = Q @ K^T ----
            v8f S[4] = {};
#pragma unroll
            for (int nt = 0; nt < 4; ++nt)
#pragma unroll
                for (int ks2 = 0; ks2 < 2; ++ks2) {
                    v16bf kfrag = frag_b16(&kt[cur][(nt * 16) * 64 + ks2 * 32], 64);
                    S[nt] = wmma_bf16(qf[ks2], kfrag, S[nt]);
                }
            // ---- causal mask on diagonal blocks ----
            if (kb * 64 + 63 > qbase) {
#pragma unroll
                for (int nt = 0; nt < 4; ++nt) {
                    int key = kb * 64 + nt * 16 + (lane & 15);
#pragma unroll
                    for (int j = 0; j < 8; ++j) {
                        int qr = qbase + j + ((lane >> 4) << 3);
                        if (key > qr) S[nt][j] = -1e30f;
                    }
                }
            }
            // ---- online softmax ----
            float corr[8];
#pragma unroll
            for (int j = 0; j < 8; ++j) {
                float m = fmaxf(fmaxf(S[0][j], S[1][j]), fmaxf(S[2][j], S[3][j]));
#pragma unroll
                for (int d = 8; d >= 1; d >>= 1)
                    m = fmaxf(m, __shfl_xor(m, d, 32));
                float nm = fmaxf(rmax[j], m);
                corr[j] = __expf(rmax[j] - nm);
                rmax[j] = nm;
            }
            ushort_t* pw = pt[wave];
            float bsum[8] = {};
#pragma unroll
            for (int nt = 0; nt < 4; ++nt)
#pragma unroll
                for (int j = 0; j < 8; ++j) {
                    float p = __expf(S[nt][j] - rmax[j]);
                    bsum[j] += p;
                    pw[(j + ((lane >> 4) << 3)) * 64 + nt * 16 + (lane & 15)] = f2bf(p);
                }
#pragma unroll
            for (int j = 0; j < 8; ++j) {
                float s = bsum[j];
#pragma unroll
                for (int d = 8; d >= 1; d >>= 1)
                    s += __shfl_xor(s, d, 32);
                rsum[j] = rsum[j] * corr[j] + s;
            }
#pragma unroll
            for (int nt = 0; nt < 4; ++nt)
#pragma unroll
                for (int j = 0; j < 8; ++j)
                    O[nt][j] *= corr[j];
            // ---- O += P @ V ----
#pragma unroll
            for (int ks2 = 0; ks2 < 2; ++ks2) {
                v16bf pf = frag_b16(pw + ks2 * 32, 64);
#pragma unroll
                for (int nt = 0; nt < 4; ++nt) {
                    v16bf vf = frag_b16(&vt[cur][(nt * 16) * 64 + ks2 * 32], 64);
                    O[nt] = wmma_bf16(pf, vf, O[nt]);
                }
            }
        }

        if (has_next) {   // transpose-scatter next V tile into the idle buffer
#pragma unroll
            for (int i = 0; i < 16; ++i) vt[nxt][(c0 + i) * 64 + r] = vv.s[i];
        }
#if HAVE_TDM
        if (wave == 0) tdm_wait();
#endif
        __syncthreads();
    }

    // ---- normalize + store ctx bf16 [B,T,DMODEL] ----
    float rinv[8];
#pragma unroll
    for (int j = 0; j < 8; ++j) rinv[j] = 1.0f / rsum[j];
#pragma unroll
    for (int nt = 0; nt < 4; ++nt)
#pragma unroll
        for (int j = 0; j < 8; ++j) {
            int token = qbase + j + ((lane >> 4) << 3);
            size_t a = ((size_t)(b * SEQ + token)) * DMODEL +
                       h * DHEAD + nt * 16 + (lane & 15);
            ctx[a] = f2bf(O[nt][j] * rinv[j]);
        }
}

// ---------------------------------------------------------------------------
extern "C" void kernel_launch(void* const* d_in, const int* in_sizes, int n_in,
                              void* d_out, int out_size, void* d_ws, size_t ws_size,
                              hipStream_t stream) {
    (void)in_sizes; (void)n_in; (void)out_size; (void)ws_size;

    const float* x  = (const float*)d_in[0];
    const float* Wq = (const float*)d_in[1];
    const float* bq = (const float*)d_in[2];
    const float* Wk = (const float*)d_in[3];
    const float* bk = (const float*)d_in[4];
    const float* Wv = (const float*)d_in[5];
    const float* bv = (const float*)d_in[6];
    const float* Wo = (const float*)d_in[7];
    const float* bo = (const float*)d_in[8];
    float* out = (float*)d_out;

    const size_t elems = (size_t)BATCH * SEQ * DMODEL;   // 8M
    const size_t wel   = (size_t)DMODEL * DMODEL;        // 1M
    ushort_t* qh  = (ushort_t*)d_ws;
    ushort_t* kh  = qh + elems;
    ushort_t* vh  = kh + elems;
    ushort_t* ctx = vh + elems;
    ushort_t* xb  = ctx + elems;
    ushort_t* wqb = xb + elems;
    ushort_t* wkb = wqb + wel;
    ushort_t* wvb = wkb + wel;
    ushort_t* wob = wvb + wel;                           // total 88 MB

    dim3 blk(256);
    cvt_kernel<<<4096, blk, 0, stream>>>(x,  xb,  (int)elems);
    cvt_kernel<<<1024, blk, 0, stream>>>(Wq, wqb, (int)wel);
    cvt_kernel<<<1024, blk, 0, stream>>>(Wk, wkb, (int)wel);
    cvt_kernel<<<1024, blk, 0, stream>>>(Wv, wvb, (int)wel);
    cvt_kernel<<<1024, blk, 0, stream>>>(Wo, wob, (int)wel);

    dim3 gg(BATCH * SEQ / 128, DMODEL / 64);             // (64,16)
    gemm_kernel<0><<<gg, blk, 0, stream>>>(xb, wqb, bq, qh, nullptr, 0.125f);
    gemm_kernel<0><<<gg, blk, 0, stream>>>(xb, wkb, bk, kh, nullptr, 1.0f);
    gemm_kernel<1><<<gg, blk, 0, stream>>>(xb, wvb, bv, vh, nullptr, 1.0f);

    attn_kernel<<<dim3(SEQ / 128, BATCH * NHEADS), blk, 0, stream>>>(qh, kh, vh, ctx);

    gemm_kernel<2><<<gg, blk, 0, stream>>>(ctx, wob, bo, nullptr, out, 1.0f);
}
